// RSSM_13735305412756
// MI455X (gfx1250) — compile-verified, matrix operand order checked by reading, not run on hardware
//
#include <hip/hip_runtime.h>
#include <math.h>

typedef __bf16 bf16;
typedef __attribute__((ext_vector_type(16))) __bf16 v16bf;
typedef __attribute__((ext_vector_type(8)))  float  v8f;

#define T_STEPS  64
#define B_TOT    256
#define OBS_D    1024
#define ACT_D    12
#define D_D      512
#define S_D      64
#define H_D      256
#define KX       96      // padded GRU-input K (64 stoch + 12 act + 20 zero pad)
#define OUT_C    704     // 3*S + D
#define M_TILE   16
#define NTHREADS 512
#define NWAVE    16

// ---------------- math helpers ----------------
__device__ __forceinline__ float sigm(float x)      { return 1.0f / (1.0f + __expf(-x)); }
__device__ __forceinline__ float eluf(float x)      { return x > 0.0f ? x : __expf(x) - 1.0f; }
__device__ __forceinline__ float softplusf(float x) { return x > 20.0f ? x : log1pf(__expf(x)); }

// ---------------- WMMA fragment helpers ----------------
// A/B bf16 fragment (16x32 / 32x16): lane L holds row (L&15);
// lanes 0-15: K = k0+0..7 and k0+16..23 ; lanes 16-31: K = k0+8..15 and k0+24..31.
__device__ __forceinline__ v16bf load_frag(const bf16* rowptr, int k0, int lane) {
  const bf16* p = rowptr + k0 + ((lane & 16) ? 8 : 0);
  v16bf f;
#pragma unroll
  for (int i = 0; i < 8; ++i) f[i] = p[i];
#pragma unroll
  for (int i = 0; i < 8; ++i) f[8 + i] = p[16 + i];
  return f;
}

__device__ __forceinline__ v8f wmma_bf16(v16bf a, v16bf b, v8f c) {
  return __builtin_amdgcn_wmma_f32_16x16x32_bf16(false, a, false, b, (short)0, c, false, false);
}

// C(16x16) tile of  A(16xK) * W(NxK)^T , W row-major with leading dim ldw.
__device__ __forceinline__ v8f gemm_tile(const bf16* A, int lda,
                                         const bf16* W, int ldw,
                                         int nbase, int ktot, int lane) {
  v8f acc = {};
  const bf16* arow = A + (size_t)(lane & 15) * lda;
  const bf16* wrow = W + (size_t)(nbase + (lane & 15)) * ldw;
  for (int k0 = 0; k0 < ktot; k0 += 32) {
    v16bf a = load_frag(arow, k0, lane);
    v16bf b = load_frag(wrow, k0, lane);
    acc = wmma_bf16(a, b, acc);
  }
  return acc;
}

// Fused 3-gate GEMM for the GRU: one shared A fragment feeds r/z/n weight rows.
__device__ __forceinline__ void tri_gemm(const bf16* A, int lda,
                                         const bf16* W, int ldw, int j0,
                                         int ktot, int lane,
                                         v8f& gr, v8f& gz, v8f& gn) {
  const bf16* arow = A + (size_t)(lane & 15) * lda;
  const bf16* wr = W + (size_t)(0 * D_D + j0 + (lane & 15)) * ldw;
  const bf16* wz = W + (size_t)(1 * D_D + j0 + (lane & 15)) * ldw;
  const bf16* wn = W + (size_t)(2 * D_D + j0 + (lane & 15)) * ldw;
  for (int k0 = 0; k0 < ktot; k0 += 32) {
    v16bf a = load_frag(arow, k0, lane);           // loaded once, used 3x
    gr = wmma_bf16(a, load_frag(wr, k0, lane), gr);
    gz = wmma_bf16(a, load_frag(wz, k0, lane), gz);
    gn = wmma_bf16(a, load_frag(wn, k0, lane), gn);
  }
}

// ---------------- prep: fp32 -> bf16 (with optional K padding) ----------------
__global__ void cvt_pad(const float* __restrict__ src, bf16* __restrict__ dst,
                        int N, int K, int Kp) {
  int i = blockIdx.x * blockDim.x + threadIdx.x;
  if (i >= N * Kp) return;
  int n = i / Kp, k = i % Kp;
  dst[i] = (k < K) ? (bf16)src[(size_t)n * K + k] : (bf16)0.0f;
}

// ---------------- main persistent RSSM kernel ----------------
__global__ __launch_bounds__(NTHREADS, 1)
void rssm_kernel(const float* __restrict__ obs,  const float* __restrict__ act,
                 const float* __restrict__ post0,
                 const float* __restrict__ eps_p, const float* __restrict__ eps_q,
                 const float* __restrict__ b_ih,  const float* __restrict__ b_hh,
                 const float* __restrict__ prb1,  const float* __restrict__ prb2,
                 const float* __restrict__ prb3,  const float* __restrict__ pob1,
                 const float* __restrict__ pob2,  const float* __restrict__ pob3,
                 const bf16* __restrict__ Wih, const bf16* __restrict__ Whh,
                 const bf16* __restrict__ pW1, const bf16* __restrict__ pW2,
                 const bf16* __restrict__ pW3, const bf16* __restrict__ qW1,
                 const bf16* __restrict__ qW2, const bf16* __restrict__ qW3,
                 float* __restrict__ posts, float* __restrict__ priors)
{
  __shared__ bf16  sPost[M_TILE * 1536];   // [ obs(1024) | deter(512) ] bf16
  __shared__ bf16  sX[M_TILE * KX];        // [ stoch(64) | act(12) | pad ] bf16
  __shared__ bf16  sH1[M_TILE * H_D];      // prior hidden 1
  __shared__ bf16  sH2[M_TILE * H_D];      // prior hidden 2
  __shared__ bf16  sG1[M_TILE * H_D];      // posterior hidden 1
  __shared__ bf16  sG2[M_TILE * H_D];      // posterior hidden 2
  __shared__ float sDet[M_TILE * D_D];     // new deter, fp32 staging

  const int tid  = threadIdx.x;
  const int lane = tid & 31;
  const int wave = tid >> 5;
  const int b0   = blockIdx.x * M_TILE;
  const int col  = lane & 15;
  const int hi8  = (lane & 16) ? 8 : 0;

  // ---- init from post0 ----
  for (int i = tid; i < M_TILE * KX; i += NTHREADS) sX[i] = (bf16)0.0f;
  __syncthreads();
  for (int i = tid; i < M_TILE * S_D; i += NTHREADS) {
    int m = i / S_D, c = i % S_D;
    sX[m * KX + c] = (bf16)post0[(size_t)(b0 + m) * OUT_C + 2 * S_D + c];
  }
  for (int i = tid; i < M_TILE * D_D; i += NTHREADS) {
    int m = i / D_D, c = i % D_D;
    sPost[m * 1536 + OBS_D + c] = (bf16)post0[(size_t)(b0 + m) * OUT_C + 3 * S_D + c];
  }
  __syncthreads();

  for (int t = 0; t < T_STEPS; ++t) {
    const size_t rowbase = (size_t)t * B_TOT + b0;

    // ---- Phase A: stage obs[t] and act[t] into LDS (bf16) ----
    for (int i = tid; i < M_TILE * OBS_D; i += NTHREADS) {
      int m = i / OBS_D, c = i % OBS_D;
      sPost[m * 1536 + c] = (bf16)obs[(rowbase + m) * OBS_D + c];
    }
    for (int i = tid; i < M_TILE * ACT_D; i += NTHREADS) {
      int m = i / ACT_D, c = i % ACT_D;
      sX[m * KX + S_D + c] = (bf16)act[(rowbase + m) * ACT_D + c];
    }
    // prefetch next step's obs tile (16 rows x 4KB = 512 cache lines; one per thread)
    if (t + 1 < T_STEPS) {
      const float* nobs = obs + (rowbase + B_TOT) * OBS_D;
      const int m = tid >> 5, off = (tid & 31) * 32;  // 32 floats = 128B line
      __builtin_prefetch(nobs + (size_t)m * OBS_D + off, 0, 3);
    }
    __syncthreads();

    // ---- Phase B: GRU, each wave owns N-tiles of the 512 gate columns ----
    for (int nt = wave; nt < D_D / 16; nt += NWAVE) {
      const int j0 = nt * 16;
      v8f ir = {}, iz = {}, in_ = {};
      tri_gemm(sX, KX, Wih, KX, j0, KX, lane, ir, iz, in_);
      v8f hr = {}, hz = {}, hn = {};
      tri_gemm(sPost + OBS_D, 1536, Whh, D_D, j0, D_D, lane, hr, hz, hn);
      const int c = j0 + col;
      const float bi_r = b_ih[c], bi_z = b_ih[D_D + c], bi_n = b_ih[2 * D_D + c];
      const float bh_r = b_hh[c], bh_z = b_hh[D_D + c], bh_n = b_hh[2 * D_D + c];
#pragma unroll
      for (int v = 0; v < 8; ++v) {
        const int m = v + hi8;
        float hold = (float)sPost[m * 1536 + OBS_D + c];
        float r = sigm(ir[v] + bi_r + hr[v] + bh_r);
        float z = sigm(iz[v] + bi_z + hz[v] + bh_z);
        float n = tanhf(in_[v] + bi_n + r * (hn[v] + bh_n));
        float h = (1.0f - z) * n + z * hold;
        sDet[m * D_D + c] = h;
        posts [(rowbase + m) * OUT_C + 3 * S_D + c] = h;
        priors[(rowbase + m) * OUT_C + 3 * S_D + c] = h;
      }
    }
    __syncthreads();

    // ---- Phase B2: re-encode new deter to bf16 in sPost ----
    for (int i = tid; i < M_TILE * D_D; i += NTHREADS) {
      int m = i / D_D, c = i % D_D;
      sPost[m * 1536 + OBS_D + c] = (bf16)sDet[i];
    }
    __syncthreads();

    // ---- Phase C: layer-1 of prior (K=512) and posterior (K=1536) ----
    for (int task = wave; task < 32; task += NWAVE) {
      if (task < 16) {
        const int n0 = task * 16;
        v8f a = gemm_tile(sPost + OBS_D, 1536, pW1, D_D, n0, D_D, lane);
        const float bb = prb1[n0 + col];
#pragma unroll
        for (int v = 0; v < 8; ++v)
          sH1[(v + hi8) * H_D + n0 + col] = (bf16)eluf(a[v] + bb);
      } else {
        const int n0 = (task - 16) * 16;
        v8f a = gemm_tile(sPost, 1536, qW1, 1536, n0, 1536, lane);
        const float bb = pob1[n0 + col];
#pragma unroll
        for (int v = 0; v < 8; ++v)
          sG1[(v + hi8) * H_D + n0 + col] = (bf16)eluf(a[v] + bb);
      }
    }
    __syncthreads();

    // ---- Phase D: layer-2 (K=256) ----
    for (int task = wave; task < 32; task += NWAVE) {
      if (task < 16) {
        const int n0 = task * 16;
        v8f a = gemm_tile(sH1, H_D, pW2, H_D, n0, H_D, lane);
        const float bb = prb2[n0 + col];
#pragma unroll
        for (int v = 0; v < 8; ++v)
          sH2[(v + hi8) * H_D + n0 + col] = (bf16)eluf(a[v] + bb);
      } else {
        const int n0 = (task - 16) * 16;
        v8f a = gemm_tile(sG1, H_D, qW2, H_D, n0, H_D, lane);
        const float bb = pob2[n0 + col];
#pragma unroll
        for (int v = 0; v < 8; ++v)
          sG2[(v + hi8) * H_D + n0 + col] = (bf16)eluf(a[v] + bb);
      }
    }
    __syncthreads();

    // ---- Phase E: output heads (paired mu/std tiles), 8 tasks on waves 0-7 ----
    if (wave < 8) {
      const int  task   = wave;
      const bool isPost = task >= 4;
      const int  n0     = (task & 3) * 16;
      const bf16*  Asrc = isPost ? sG2  : sH2;
      const bf16*  W3   = isPost ? qW3  : pW3;
      const float* b3   = isPost ? pob3 : prb3;
      const float* eps  = isPost ? eps_q : eps_p;
      float*       outp = isPost ? posts : priors;
      v8f mu = gemm_tile(Asrc, H_D, W3, H_D, n0,       H_D, lane);
      v8f sd = gemm_tile(Asrc, H_D, W3, H_D, S_D + n0, H_D, lane);
      const float bmu = b3[n0 + col];
      const float bsd = b3[S_D + n0 + col];
#pragma unroll
      for (int v = 0; v < 8; ++v) {
        const int m = v + hi8;
        float muv = mu[v] + bmu;
        float sv  = sd[v] + bsd;
        sv = fmaxf(sv, -18.0f);
        sv = softplusf(sv) + 0.1f;
        float e  = eps[(rowbase + m) * S_D + n0 + col];
        float st = muv + sv * e;
        const size_t ob = (rowbase + m) * OUT_C;
        outp[ob + n0 + col]           = muv;
        outp[ob + S_D + n0 + col]     = sv;
        outp[ob + 2 * S_D + n0 + col] = st;
        if (isPost) sX[m * KX + n0 + col] = (bf16)st;  // stoch feeds next step
      }
    }
    __syncthreads();
  }
}

// ---------------- host entry ----------------
extern "C" void kernel_launch(void* const* d_in, const int* in_sizes, int n_in,
                              void* d_out, int out_size, void* d_ws, size_t ws_size,
                              hipStream_t stream) {
  (void)in_sizes; (void)n_in; (void)out_size; (void)ws_size;
  bf16* ws = (bf16*)d_ws;
  size_t o = 0;
  bf16* Wih = ws + o; o += (size_t)1536 * 96;
  bf16* Whh = ws + o; o += (size_t)1536 * 512;
  bf16* pW1 = ws + o; o += (size_t)256 * 512;
  bf16* pW2 = ws + o; o += (size_t)256 * 256;
  bf16* pW3 = ws + o; o += (size_t)128 * 256;
  bf16* qW1 = ws + o; o += (size_t)256 * 1536;
  bf16* qW2 = ws + o; o += (size_t)256 * 256;
  bf16* qW3 = ws + o; o += (size_t)128 * 256;

  auto cvt = [&](const void* src, bf16* dst, int N, int K, int Kp) {
    int n = N * Kp;
    cvt_pad<<<(n + 255) / 256, 256, 0, stream>>>((const float*)src, dst, N, K, Kp);
  };
  cvt(d_in[5],  Wih, 1536, 76,   96);    // W_ih  (padded K 76->96)
  cvt(d_in[6],  Whh, 1536, 512,  512);   // W_hh
  cvt(d_in[9],  pW1, 256,  512,  512);   // prW1
  cvt(d_in[11], pW2, 256,  256,  256);   // prW2
  cvt(d_in[13], pW3, 128,  256,  256);   // prW3
  cvt(d_in[15], qW1, 256,  1536, 1536);  // poW1
  cvt(d_in[17], qW2, 256,  256,  256);   // poW2
  cvt(d_in[19], qW3, 128,  256,  256);   // poW3

  float* posts  = (float*)d_out;
  float* priors = posts + (size_t)T_STEPS * B_TOT * OUT_C;

  rssm_kernel<<<B_TOT / M_TILE, NTHREADS, 0, stream>>>(
      (const float*)d_in[0],  // observations
      (const float*)d_in[1],  // actions
      (const float*)d_in[2],  // post0
      (const float*)d_in[3],  // noise_prior
      (const float*)d_in[4],  // noise_post
      (const float*)d_in[7],  // b_ih
      (const float*)d_in[8],  // b_hh
      (const float*)d_in[10], // prb1
      (const float*)d_in[12], // prb2
      (const float*)d_in[14], // prb3
      (const float*)d_in[16], // pob1
      (const float*)d_in[18], // pob2
      (const float*)d_in[20], // pob3
      Wih, Whh, pW1, pW2, pW3, qW1, qW2, qW3,
      posts, priors);
}
